// RecurrentLSTM_17583596110128
// MI455X (gfx1250) — compile-verified, matrix operand order checked by reading, not run on hardware
//
#include <hip/hip_runtime.h>
#include <hip/hip_bf16.h>
#include <math.h>

// ---------------- constants ----------------
#define BB   64      // batch
#define TT   1024    // input time
#define WW   128     // input width (conv in-channels)
#define CIN  256     // conv out-channels (LSTM input size)
#define HH   512     // LSTM hidden
#define NC   25      // classes
#define KK   65      // conv kernel taps
#define LL   960     // conv valid length
#define POOL 15      // pool window
#define TP   64      // pooled timesteps
#define FOURH 2048   // 4*H

typedef __attribute__((ext_vector_type(16))) __bf16 v16bf;
typedef __attribute__((ext_vector_type(8)))  float  v8f;
typedef unsigned int u32x4 __attribute__((ext_vector_type(4)));

__device__ __forceinline__ __bf16 f2bf(float x) {
    unsigned u = __float_as_uint(x);
    unsigned r = (u + 0x7fffu + ((u >> 16) & 1u)) >> 16;
    unsigned short s = (unsigned short)r;
    __bf16 b;
    __builtin_memcpy(&b, &s, 2);
    return b;
}

// Load a 16x32 bf16 WMMA fragment whose per-lane rows (m = lane&15) are
// contiguous in memory with stride `rowStride` (in halves), starting at
// column k0. Matches CDNA5 16-bit A/B VGPR layout:
//   half h of lane L holds K = (h<8 ? h : h+8) + (L>=16 ? 8 : 0) + k0.
__device__ __forceinline__ v16bf load_frag(const __bf16* __restrict__ base,
                                           int rowStride, int k0) {
    const int lane = threadIdx.x & 31;
    const int m   = lane & 15;
    const int hi8 = (lane >> 4) << 3;          // 0 or 8
    const __bf16* p = base + (size_t)m * rowStride + k0 + hi8;
    union { u32x4 q[2]; v16bf v; } u;
    u.q[0] = *(const u32x4*)(p);
    u.q[1] = *(const u32x4*)(p + 16);
    return u.v;
}

// 2x2 register-blocked WMMA step: A rows [0..32) x K32 at k0, Bt rows [0..32).
// 4 fragment loads feed 4 WMMAs -> 2x the FLOP/byte of the unblocked form.
__device__ __forceinline__ void wmma2x2(const __bf16* __restrict__ A, int as,
                                        const __bf16* __restrict__ Bt, int bs,
                                        int k0, v8f acc[4]) {
    v16bf a0 = load_frag(A, as, k0);
    v16bf a1 = load_frag(A + (size_t)16 * as, as, k0);
    v16bf b0 = load_frag(Bt, bs, k0);
    v16bf b1 = load_frag(Bt + (size_t)16 * bs, bs, k0);
    acc[0] = __builtin_amdgcn_wmma_f32_16x16x32_bf16(false, a0, false, b0,
                                                     (short)0, acc[0], false, false);
    acc[1] = __builtin_amdgcn_wmma_f32_16x16x32_bf16(false, a0, false, b1,
                                                     (short)0, acc[1], false, false);
    acc[2] = __builtin_amdgcn_wmma_f32_16x16x32_bf16(false, a1, false, b0,
                                                     (short)0, acc[2], false, false);
    acc[3] = __builtin_amdgcn_wmma_f32_16x16x32_bf16(false, a1, false, b1,
                                                     (short)0, acc[3], false, false);
}

// ---------------- conversion kernels ----------------
__global__ __launch_bounds__(256) void cvt_f32_bf16(const float* __restrict__ src,
                                                    __bf16* __restrict__ dst, int n) {
    int i = blockIdx.x * 256 + threadIdx.x;
    if (i < n) dst[i] = f2bf(src[i]);
}

// conv_w [CIN][WW][KK] f32  ->  wh [KK][CIN][WW] bf16  (per-tap row-major)
__global__ __launch_bounds__(256) void cvt_conv_w(const float* __restrict__ cw,
                                                  __bf16* __restrict__ wh) {
    int i = blockIdx.x * 256 + threadIdx.x;
    if (i >= KK * CIN * WW) return;
    int w   = i & (WW - 1);
    int cin = (i >> 7) & (CIN - 1);
    int k   = i >> 15;
    wh[i] = f2bf(cw[((size_t)cin * WW + w) * KK + k]);
}

__global__ __launch_bounds__(256) void init_state(const float* __restrict__ h0,
                                                  const float* __restrict__ c0,
                                                  float* __restrict__ hf,
                                                  __bf16* __restrict__ hb,
                                                  float* __restrict__ c) {
    int i = blockIdx.x * 256 + threadIdx.x;
    if (i >= BB * HH) return;
    float h = h0[i];
    hf[i] = h;
    hb[i] = f2bf(h);
    c[i]  = c0[i];
}

// ---------------- conv1d as 65 tap-GEMMs, output-transposed ----------------
// out_t[b][l][cin] = sum_k sum_w xh[b][l+k][w] * wh[k][cin][w]
// grid: (b * 15 + mBlk) * 4 + nBlk; 128 threads = 4 waves in a 2x2 grid.
// Each WG computes a 64(l) x 64(cin) tile; each wave a 32x32 sub-tile with
// 4 accumulators (2x2 register blocking).
__global__ __launch_bounds__(128) void conv_gemm_kernel(
        const __bf16* __restrict__ xh,       // [BB][TT][WW]
        const __bf16* __restrict__ wh,       // [KK][CIN][WW]
        float* __restrict__ convout) {       // [BB][LL][CIN]
    const int bi   = blockIdx.x;
    const int nBlk = bi & 3;
    const int mBlk = (bi >> 2) % 15;
    const int b    = bi / 60;
    const int wave  = threadIdx.x >> 5;
    const int waveM = wave >> 1;
    const int waveN = wave & 1;
    const int l0   = mBlk * 64 + waveM * 32;
    const int cin0 = nBlk * 64 + waveN * 32;

    v8f acc[4] = {};
    for (int k = 0; k < KK; ++k) {
        const __bf16* A  = xh + ((size_t)b * TT + l0 + k) * WW;
        const __bf16* Bt = wh + ((size_t)k * CIN + cin0) * WW;
        // hide latency of the sliding-window A rows (emits global_prefetch)
        __builtin_prefetch(A + 8 * WW, 0, 0);
#pragma unroll
        for (int w0 = 0; w0 < WW; w0 += 32)
            wmma2x2(A, WW, Bt, WW, w0, acc);
    }

    const int lane = threadIdx.x & 31;
    const int hi8  = (lane >> 4) << 3;
    const int col  = lane & 15;
    float* outp = convout + ((size_t)b * LL + l0) * CIN + cin0;
#pragma unroll
    for (int t = 0; t < 4; ++t) {
        const int mo = (t >> 1) * 16, no = (t & 1) * 16;
#pragma unroll
        for (int r = 0; r < 8; ++r)
            outp[(size_t)(mo + r + hi8) * CIN + no + col] = acc[t][r];
    }
}

// ---------------- bias + adaptive max pool(15) -> bf16 sequence -------------
__global__ __launch_bounds__(256) void pool_kernel(
        const float* __restrict__ convout,   // [BB][LL][CIN]
        const float* __restrict__ conv_b,    // [CIN]
        __bf16* __restrict__ seq) {          // [TP][BB][CIN]
    int idx = blockIdx.x * 256 + threadIdx.x;    // TP*BB*CIN = 1,048,576
    if (idx >= TP * BB * CIN) return;
    const int cin = idx & (CIN - 1);
    const int b   = (idx >> 8) & (BB - 1);
    const int tp  = idx >> 14;
    const float* p = convout + ((size_t)b * LL + tp * POOL) * CIN + cin;
    float mx = -3.4e38f;
#pragma unroll
    for (int j = 0; j < POOL; ++j) mx = fmaxf(mx, p[(size_t)j * CIN]);
    seq[idx] = f2bf(mx + conv_b[cin]);
}

// ---------------- LSTM gates GEMM: [64 x 2048] = xt*w_ih^T + h*w_hh^T -------
// grid: 32 blocks x 128 threads; one 32x32 (2x2-blocked) tile per wave.
__global__ __launch_bounds__(128) void lstm_gates_kernel(
        const __bf16* __restrict__ xt,    // [BB][CIN]
        const __bf16* __restrict__ hbf,   // [BB][HH]
        const __bf16* __restrict__ wih,   // [4H][CIN] (== B^T layout)
        const __bf16* __restrict__ whh,   // [4H][HH]
        float* __restrict__ gates) {      // [BB][4H]
    const int wave   = threadIdx.x >> 5;
    const int tileId = blockIdx.x * 4 + wave;   // 0..127
    const int m0 = (tileId & 1) * 32;           // 64 / 32
    const int n0 = (tileId >> 1) * 32;          // 2048 / 32

    v8f acc[4] = {};
    const __bf16* A1 = xt  + (size_t)m0 * CIN;
    const __bf16* B1 = wih + (size_t)n0 * CIN;
#pragma unroll
    for (int k0 = 0; k0 < CIN; k0 += 32)
        wmma2x2(A1, CIN, B1, CIN, k0, acc);
    const __bf16* A2 = hbf + (size_t)m0 * HH;
    const __bf16* B2 = whh + (size_t)n0 * HH;
#pragma unroll
    for (int k0 = 0; k0 < HH; k0 += 32)
        wmma2x2(A2, HH, B2, HH, k0, acc);

    const int lane = threadIdx.x & 31;
    const int hi8  = (lane >> 4) << 3;
    const int col  = lane & 15;
#pragma unroll
    for (int t = 0; t < 4; ++t) {
        const int mo = (t >> 1) * 16, no = (t & 1) * 16;
#pragma unroll
        for (int r = 0; r < 8; ++r)
            gates[(size_t)(m0 + mo + r + hi8) * FOURH + n0 + no + col] = acc[t][r];
    }
}

__global__ __launch_bounds__(256) void lstm_pointwise_kernel(
        const float* __restrict__ gates, const float* __restrict__ b_ih,
        const float* __restrict__ b_hh, float* __restrict__ c,
        float* __restrict__ hf, __bf16* __restrict__ hb) {
    int idx = blockIdx.x * 256 + threadIdx.x;
    if (idx >= BB * HH) return;
    const int b  = idx >> 9;
    const int hI = idx & (HH - 1);
    const float* gr = gates + (size_t)b * FOURH;
    float gi = gr[hI]          + b_ih[hI]          + b_hh[hI];
    float gf = gr[HH + hI]     + b_ih[HH + hI]     + b_hh[HH + hI];
    float gg = gr[2 * HH + hI] + b_ih[2 * HH + hI] + b_hh[2 * HH + hI];
    float go = gr[3 * HH + hI] + b_ih[3 * HH + hI] + b_hh[3 * HH + hI];
    float si = 1.f / (1.f + __expf(-gi));
    float sf = 1.f / (1.f + __expf(-gf));
    float so = 1.f / (1.f + __expf(-go));
    float cn = sf * c[idx] + si * tanhf(gg);
    c[idx] = cn;
    float h = so * tanhf(cn);
    hf[idx] = h;
    hb[idx] = f2bf(h);
}

// ---------------- FC + softmax ----------------
__global__ __launch_bounds__(32) void fc_softmax_kernel(
        const float* __restrict__ hf, const float* __restrict__ fcw,
        const float* __restrict__ fcb, float* __restrict__ out) {
    __shared__ float sl[32];
    const int b = blockIdx.x;
    const int n = threadIdx.x;
    float v = -3.4e38f;
    if (n < NC) {
        float d = fcb[n];
        const float* hr = hf  + (size_t)b * HH;
        const float* wr = fcw + (size_t)n * HH;
        for (int k = 0; k < HH; ++k) d += hr[k] * wr[k];
        v = d;
    }
    sl[n] = v;
    __syncthreads();
    if (n < NC) {
        float mx = -3.4e38f;
        for (int j = 0; j < NC; ++j) mx = fmaxf(mx, sl[j]);
        float s = 0.f;
        for (int j = 0; j < NC; ++j) s += __expf(sl[j] - mx);
        out[b * NC + n] = __expf(sl[n] - mx) / s;
    }
}

// ---------------- host ----------------
extern "C" void kernel_launch(void* const* d_in, const int* in_sizes, int n_in,
                              void* d_out, int out_size, void* d_ws, size_t ws_size,
                              hipStream_t stream) {
    const float* input_ = (const float*)d_in[0];
    const float* h0     = (const float*)d_in[1];
    const float* c0     = (const float*)d_in[2];
    const float* conv_w = (const float*)d_in[3];
    const float* conv_b = (const float*)d_in[4];
    const float* w_ih   = (const float*)d_in[5];
    const float* w_hh   = (const float*)d_in[6];
    const float* b_ih   = (const float*)d_in[7];
    const float* b_hh   = (const float*)d_in[8];
    const float* fc_w   = (const float*)d_in[9];
    const float* fc_b   = (const float*)d_in[10];
    float* out = (float*)d_out;

    char* ws = (char*)d_ws;
    size_t off = 0;
    auto alloc = [&](size_t bytes) -> void* {
        void* p = ws + off;
        off = (off + bytes + 255) & ~(size_t)255;
        return p;
    };
    __bf16* xh      = (__bf16*)alloc((size_t)BB * TT * WW * 2);    // 16.8 MB
    __bf16* wh      = (__bf16*)alloc((size_t)KK * CIN * WW * 2);   // 4.3 MB
    __bf16* wihb    = (__bf16*)alloc((size_t)FOURH * CIN * 2);     // 1.0 MB
    __bf16* whhb    = (__bf16*)alloc((size_t)FOURH * HH * 2);      // 2.1 MB
    __bf16* seq     = (__bf16*)alloc((size_t)TP * BB * CIN * 2);   // 2.1 MB
    float*  convout = (float*) alloc((size_t)BB * LL * CIN * 4);   // 62.9 MB
    float*  gates   = (float*) alloc((size_t)BB * FOURH * 4);      // 0.5 MB
    float*  hf      = (float*) alloc((size_t)BB * HH * 4);
    __bf16* hb      = (__bf16*)alloc((size_t)BB * HH * 2);
    float*  cbuf    = (float*) alloc((size_t)BB * HH * 4);

    // conversions
    {
        int n = BB * TT * WW;
        cvt_f32_bf16<<<(n + 255) / 256, 256, 0, stream>>>(input_, xh, n);
    }
    cvt_conv_w<<<(KK * CIN * WW + 255) / 256, 256, 0, stream>>>(conv_w, wh);
    {
        int n = FOURH * CIN;
        cvt_f32_bf16<<<(n + 255) / 256, 256, 0, stream>>>(w_ih, wihb, n);
    }
    {
        int n = FOURH * HH;
        cvt_f32_bf16<<<(n + 255) / 256, 256, 0, stream>>>(w_hh, whhb, n);
    }
    init_state<<<(BB * HH + 255) / 256, 256, 0, stream>>>(h0, c0, hf, hb, cbuf);

    // conv GEMM -> f32 scratch, then fused bias+pool -> bf16 sequence
    conv_gemm_kernel<<<BB * 15 * 4, 128, 0, stream>>>(xh, wh, convout);
    pool_kernel<<<(TP * BB * CIN + 255) / 256, 256, 0, stream>>>(convout, conv_b, seq);

    // LSTM: 64 timesteps
    for (int t = 0; t < TP; ++t) {
        const __bf16* xt = seq + (size_t)t * BB * CIN;
        lstm_gates_kernel<<<32, 128, 0, stream>>>(xt, hb, wihb, whhb, gates);
        lstm_pointwise_kernel<<<(BB * HH + 255) / 256, 256, 0, stream>>>(
                gates, b_ih, b_hh, cbuf, hf, hb);
    }

    // FC + softmax on final h
    fc_softmax_kernel<<<BB, 32, 0, stream>>>(hf, fc_w, fc_b, out);
    (void)in_sizes; (void)n_in; (void)out_size; (void)ws_size;
}
